// RNNLanguageModel_44822278701460
// MI455X (gfx1250) — matrix operand order repaired and checked by reading, compiled, not run on hardware
//
#include <hip/hip_runtime.h>
#include <hip/hip_bf16.h>
#include <math.h>

// ---------------------------------------------------------------------------
// Problem constants
// ---------------------------------------------------------------------------
#define Bsz 16
#define Tfull 257
#define Tm 256
#define Vv 32000
#define Ee 256
#define Hh 512
#define G4H 2048          // 4*H
#define NBLK_RNN 32       // cooperative workgroups, one per WGP

typedef __attribute__((ext_vector_type(16))) __bf16 v16bf;
typedef __attribute__((ext_vector_type(8)))  __bf16 v8bf;
typedef __attribute__((ext_vector_type(8)))  float  v8f;

__device__ __forceinline__ v8f wmma_bf16(v16bf a, v16bf b, v8f c) {
  // D = A(16x32) * B(32x16) + C, f32 accumulate
  return __builtin_amdgcn_wmma_f32_16x16x32_bf16(false, a, false, b,
                                                 (short)0, c, false, false);
}

__device__ __forceinline__ float sigm(float x) { return 1.f / (1.f + expf(-x)); }

// A-fragment (16x32 bf16) loader: lane l -> row m=l&15, half kh=l>>4.
// elements 0..15 = K (kb + kh*16 .. +15), contiguous (matches B fill)
__device__ __forceinline__ v16bf load_a_frag(const __bf16* rowp, int kloc, int kh) {
  v16bf a;
  ((v8bf*)&a)[0] = *(const v8bf*)(rowp + kloc + kh * 16);
  ((v8bf*)&a)[1] = *(const v8bf*)(rowp + kloc + kh * 16 + 8);
  return a;
}

// ---------------------------------------------------------------------------
// Workspace layout (bytes)
// ---------------------------------------------------------------------------
#define WS_BAR      0                         // grid barrier (16 ints)
#define WS_H0G      256                       // 2 * 16*512 bf16 (double buffer)
#define WS_H1G      33024
#define WS_ZERO_END 66048                     // region zeroed each launch
#define WS_XBF      66048                     // [Tm][B][E] bf16    (2 MB)
#define WS_K0T      2163200                   // [2048][768]  bf16  (3 MB)
#define WS_K1T      5308928                   // [2048][1024] bf16  (4 MB)
#define WS_WPT      9503232                   // [32000][512] bf16  (32.8 MB)
#define WS_RNN      42271232                  // [4096][512]  bf16  (4 MB)
#define WS_LSE      46465536                  // [4096] f32
#define WS_CE       46481920                  // [4096] f32

// rnn kernel dynamic LDS partition (bytes)
#define LDS_W0   0                            // 4*16*768  bf16 = 98304
#define LDS_W1   98304                        // 4*16*1024 bf16 = 131072
#define LDS_Z    229376                       // 4*16*16   f32  = 4096
#define LDS_C0   233472                       // 16*16     f32  = 1024
#define LDS_C1   234496                       // 16*16     f32  = 1024
#define RNN_LDS_BYTES 235520

// ---------------------------------------------------------------------------
// TDM weight load (amdgpu-toolchain clang-23: 6-arg builtin), manual fallback
// ---------------------------------------------------------------------------
#if __has_builtin(__builtin_amdgcn_tensor_load_to_lds)
#define HAVE_TDM 1
typedef __attribute__((ext_vector_type(4))) unsigned int u32x4;
typedef __attribute__((ext_vector_type(8))) int          i32x8;
typedef __attribute__((ext_vector_type(4))) int          i32x4;

__device__ __forceinline__ void tdm_load_3d(unsigned lds_off, const void* gaddr,
                                            unsigned tile_d0, unsigned tile_d1,
                                            unsigned tile_d2,
                                            unsigned long long dim0_stride,
                                            unsigned long long dim1_stride,
                                            unsigned tdim0, unsigned tdim1,
                                            unsigned tdim2) {
  unsigned long long ga = (unsigned long long)gaddr;
  u32x4 g0 = {
      1u,                                            // count=1 (valid user D#)
      lds_off,                                       // lds_addr
      (unsigned)(ga & 0xFFFFFFFFu),                  // global_addr[31:0]
      (unsigned)((ga >> 32) & 0x1FFFFFFu) | (2u << 30)  // addr[56:32] | type=2
  };
  i32x8 g1 = {
      (int)(1u << 16),                               // data_size=1 (2 bytes)
      (int)((tdim0 & 0xFFFFu) << 16),                // tensor_dim0[15:0]
      (int)((tdim0 >> 16) | ((tdim1 & 0xFFFFu) << 16)),
      (int)((tdim1 >> 16) | (tile_d0 << 16)),        // dim1 hi | tile_dim0
      (int)(tile_d1 | (tile_d2 << 16)),              // tile_dim1 | tile_dim2
      (int)(dim0_stride & 0xFFFFFFFFu),
      (int)(((dim0_stride >> 32) & 0xFFFFu) | ((dim1_stride & 0xFFFFu) << 16)),
      (int)((dim1_stride >> 16) & 0xFFFFFFFFull)
  };
  i32x4 g2 = { (int)tdim2, 0, 0, 0 };                // tensor_dim2; rest unused
  i32x4 g3 = { 0, 0, 0, 0 };
  i32x8 g4 = { 0, 0, 0, 0, 0, 0, 0, 0 };             // extra group (clang-23 form)
  __builtin_amdgcn_tensor_load_to_lds(g0, g1, g2, g3, g4, 0);
}
#else
#define HAVE_TDM 0
#endif

__device__ __forceinline__ void wait_tensor0() {
#if __has_builtin(__builtin_amdgcn_s_wait_tensorcnt)
  __builtin_amdgcn_s_wait_tensorcnt(0);
#else
  asm volatile("s_wait_tensorcnt 0" ::: "memory");
#endif
}

// ---------------------------------------------------------------------------
// Grid-wide barrier (cooperative; all NBLK_RNN blocks resident)
// ---------------------------------------------------------------------------
__device__ __forceinline__ void grid_sync(int* bar, int nblk, int& gen) {
  __threadfence();
  __syncthreads();
  if (threadIdx.x == 0) {
    gen += nblk;
    __hip_atomic_fetch_add(bar, 1, __ATOMIC_ACQ_REL, __HIP_MEMORY_SCOPE_AGENT);
    while (__hip_atomic_load(bar, __ATOMIC_ACQUIRE, __HIP_MEMORY_SCOPE_AGENT) < gen) {
      __builtin_amdgcn_s_sleep(1);
    }
  }
  __syncthreads();
}

// ---------------------------------------------------------------------------
// Kernel 1: embedding gather -> bf16 x[t][b][e]
// ---------------------------------------------------------------------------
__global__ void embed_kernel(const int* __restrict__ texts,
                             const float* __restrict__ emb,
                             __bf16* __restrict__ xbf) {
  int idx = blockIdx.x * 256 + threadIdx.x;      // < Tm*B*E = 1048576
  int e = idx & (Ee - 1);
  int bt = idx >> 8;                             // t*16 + b
  int b = bt & (Bsz - 1), t = bt >> 4;
  int tok = texts[b * Tfull + t];
  xbf[idx] = (__bf16)emb[(size_t)tok * Ee + e];
}

// ---------------------------------------------------------------------------
// Kernel 2: f32 [R][C] -> bf16 transposed [C][R]
// ---------------------------------------------------------------------------
__global__ void transpose_bf16_kernel(const float* __restrict__ src,
                                      __bf16* __restrict__ dst, int R, int C) {
  long long total = (long long)R * C;
  for (long long idx = (long long)blockIdx.x * 256 + threadIdx.x; idx < total;
       idx += (long long)gridDim.x * 256) {
    int r = (int)(idx / C), c = (int)(idx % C);
    dst[(long long)c * R + r] = (__bf16)src[idx];
  }
}

// ---------------------------------------------------------------------------
// Kernel 3: cooperative 2-layer LSTM. 32 blocks x 128 threads (4 waves).
// Block owns hidden columns [n0, n0+16); wave w owns gate tile w (i/j/f/o).
// Weights resident in LDS (224 KB) via TDM; h state double-buffered in L2.
// ---------------------------------------------------------------------------
__global__ void rnn_kernel(const __bf16* __restrict__ xbf,
                           const __bf16* __restrict__ k0T,
                           const __bf16* __restrict__ k1T,
                           const float* __restrict__ b0,
                           const float* __restrict__ b1,
                           const int* __restrict__ tlen,
                           __bf16* __restrict__ h0g,   // [2][16][512]
                           __bf16* __restrict__ h1g,   // [2][16][512]
                           __bf16* __restrict__ rnnb,  // [4096][512]
                           int* __restrict__ bar) {
  extern __shared__ char smem[];
  __bf16* wlds0 = (__bf16*)(smem + LDS_W0);   // [4*16][768]
  __bf16* wlds1 = (__bf16*)(smem + LDS_W1);   // [4*16][1024]
  float*  zlds  = (float*)(smem + LDS_Z);     // [4][16][16]
  float*  clds0 = (float*)(smem + LDS_C0);    // [16][16]
  float*  clds1 = (float*)(smem + LDS_C1);    // [16][16]

  const int n0 = blockIdx.x * 16;
  const int lane = threadIdx.x & 31;
  const int wv = threadIdx.x >> 5;            // gate id 0..3
  const int m = lane & 15;                    // A-row / B-col / D-col index
  const int kh = lane >> 4;

  // ---- one-time weight residency: 2 TDM 3-D tile loads (K x 16 x 4) ----
#if HAVE_TDM
  if (threadIdx.x < 32) {
    unsigned ldsbase = (unsigned)(size_t)(void*)smem;
    tdm_load_3d(ldsbase + LDS_W0, k0T + (size_t)n0 * 768,
                768, 16, 4, 768ull, 512ull * 768ull, 768, G4H, 4);
    tdm_load_3d(ldsbase + LDS_W1, k1T + (size_t)n0 * 1024,
                1024, 16, 4, 1024ull, 512ull * 1024ull, 1024, G4H, 4);
    wait_tensor0();
  }
#else
  for (int i = threadIdx.x; i < 4 * 16 * 768; i += 128) {
    int k = i % 768, rr = i / 768, g = rr >> 4, row = rr & 15;
    wlds0[i] = k0T[(size_t)(g * 512 + n0 + row) * 768 + k];
  }
  for (int i = threadIdx.x; i < 4 * 16 * 1024; i += 128) {
    int k = i & 1023, rr = i >> 10, g = rr >> 4, row = rr & 15;
    wlds1[i] = k1T[(size_t)(g * 512 + n0 + row) * 1024 + k];
  }
#endif
  for (int i = threadIdx.x; i < 256; i += 128) { clds0[i] = 0.f; clds1[i] = 0.f; }
  __syncthreads();

  int gen = 0;
  const __bf16* wrow0 = wlds0 + (size_t)(wv * 16 + m) * 768;   // B: col m of tile
  const __bf16* wrow1 = wlds1 + (size_t)(wv * 16 + m) * 1024;

  for (int t = 0; t < Tm; ++t) {
    const int p = t & 1;
    const __bf16* h0r = h0g + p * (Bsz * Hh);
    __bf16*       h0w = h0g + (p ^ 1) * (Bsz * Hh);
    const __bf16* h1r = h1g + p * (Bsz * Hh);
    __bf16*       h1w = h1g + (p ^ 1) * (Bsz * Hh);
    const __bf16* xt  = xbf + (size_t)t * (Bsz * Ee);

    // ======== layer 0: z = [x_t | h0] @ k0, K = 768 ========
    v8f acc = {0.f, 0.f, 0.f, 0.f, 0.f, 0.f, 0.f, 0.f};
    for (int kb = 0; kb < 768; kb += 32) {
      const __bf16* rowp;
      int kloc;
      if (kb < Ee) { rowp = xt + m * Ee;  kloc = kb; }
      else         { rowp = h0r + m * Hh; kloc = kb - Ee; }
      v16bf a = load_a_frag(rowp, kloc, kh);
      v16bf b = *(const v16bf*)(wrow0 + kb + kh * 16);
      acc = wmma_bf16(a, b, acc);
    }
    {
      float* zt = zlds + wv * 256;
#pragma unroll
      for (int v = 0; v < 8; ++v) zt[(v + 8 * kh) * 16 + m] = acc[v];
    }
    __syncthreads();
    // LSTM cell, columns n0..n0+15 (128 threads x 2 rows)
    {
      int col = threadIdx.x & 15;
      int r0 = threadIdx.x >> 4;
#pragma unroll
      for (int rr = 0; rr < 2; ++rr) {
        int row = r0 + rr * 8;                         // batch index
        float iv = zlds[0 * 256 + row * 16 + col] + b0[0 * Hh + n0 + col];
        float jv = zlds[1 * 256 + row * 16 + col] + b0[1 * Hh + n0 + col];
        float fv = zlds[2 * 256 + row * 16 + col] + b0[2 * Hh + n0 + col];
        float ov = zlds[3 * 256 + row * 16 + col] + b0[3 * Hh + n0 + col];
        float c = clds0[row * 16 + col];
        float nc = c * sigm(fv + 1.f) + sigm(iv) * tanhf(jv);
        float nh = tanhf(nc) * sigm(ov);
        bool mk = t < (tlen[row] - 1);
        clds0[row * 16 + col] = mk ? nc : c;
        float hold = (float)h0r[row * Hh + n0 + col];
        h0w[row * Hh + n0 + col] = (__bf16)(mk ? nh : hold);
      }
    }
    grid_sync(bar, NBLK_RNN, gen);   // all WGPs publish new h0

    // ======== layer 1: z = [h0_new | h1] @ k1, K = 1024 ========
    acc = (v8f){0.f, 0.f, 0.f, 0.f, 0.f, 0.f, 0.f, 0.f};
    for (int kb = 0; kb < 1024; kb += 32) {
      const __bf16* rowp;
      int kloc;
      if (kb < Hh) { rowp = h0w + m * Hh; kloc = kb; }
      else         { rowp = h1r + m * Hh; kloc = kb - Hh; }
      v16bf a = load_a_frag(rowp, kloc, kh);
      v16bf b = *(const v16bf*)(wrow1 + kb + kh * 16);
      acc = wmma_bf16(a, b, acc);
    }
    {
      float* zt = zlds + wv * 256;
#pragma unroll
      for (int v = 0; v < 8; ++v) zt[(v + 8 * kh) * 16 + m] = acc[v];
    }
    __syncthreads();
    {
      int col = threadIdx.x & 15;
      int r0 = threadIdx.x >> 4;
#pragma unroll
      for (int rr = 0; rr < 2; ++rr) {
        int row = r0 + rr * 8;
        float iv = zlds[0 * 256 + row * 16 + col] + b1[0 * Hh + n0 + col];
        float jv = zlds[1 * 256 + row * 16 + col] + b1[1 * Hh + n0 + col];
        float fv = zlds[2 * 256 + row * 16 + col] + b1[2 * Hh + n0 + col];
        float ov = zlds[3 * 256 + row * 16 + col] + b1[3 * Hh + n0 + col];
        float c = clds1[row * 16 + col];
        float nc = c * sigm(fv + 1.f) + sigm(iv) * tanhf(jv);
        float nh = tanhf(nc) * sigm(ov);
        bool mk = t < (tlen[row] - 1);
        clds1[row * 16 + col] = mk ? nc : c;
        float hold = (float)h1r[row * Hh + n0 + col];
        h1w[row * Hh + n0 + col] = (__bf16)(mk ? nh : hold);
        // rnn_out = m * nh1
        rnnb[((size_t)row * Tm + t) * Hh + n0 + col] = (__bf16)(mk ? nh : 0.f);
      }
    }
    grid_sync(bar, NBLK_RNN, gen);   // all WGPs publish new h1
  }
}

// ---------------------------------------------------------------------------
// Kernel 4: logits = rnn_out @ Wp + bp. Block = 8 waves; block tile 16x128.
// A (16x512 bf16) staged in LDS once per block; B streamed from WpT.
// ---------------------------------------------------------------------------
__global__ void logits_kernel(const __bf16* __restrict__ rnn,
                              const __bf16* __restrict__ WpT,
                              const float* __restrict__ bp,
                              float* __restrict__ out) {
  __shared__ __bf16 als[16 * 520];                   // padded rows
  const int r0 = blockIdx.y * 16;
  const int tid = threadIdx.x;
  {
    int row = tid >> 4, seg = tid & 15;              // 32 bf16 per thread
    const uint4* src = (const uint4*)(rnn + (size_t)(r0 + row) * Hh + seg * 32);
    uint4* dst = (uint4*)(als + row * 520 + seg * 32);
#pragma unroll
    for (int i = 0; i < 4; ++i) dst[i] = src[i];
  }
  __syncthreads();

  const int wv = tid >> 5;
  const int lane = tid & 31, n = lane & 15, kh = lane >> 4;
  const int ncol0 = (blockIdx.x * 8 + wv) * 16;
  const __bf16* wrow = WpT + (size_t)(ncol0 + n) * Hh;
  const __bf16* arow = als + n * 520;                // A-row index = lane&15

  v8f acc = {0.f, 0.f, 0.f, 0.f, 0.f, 0.f, 0.f, 0.f};
#pragma unroll
  for (int kb = 0; kb < Hh; kb += 32) {
    __builtin_prefetch(wrow + kb + 256, 0, 1);       // global_prefetch_b8
    v16bf a = load_a_frag(arow, kb, kh);
    v16bf b = *(const v16bf*)(wrow + kb + kh * 16);
    acc = wmma_bf16(a, b, acc);
  }
  float bias = bp[ncol0 + n];
#pragma unroll
  for (int v = 0; v < 8; ++v) {
    int row = r0 + v + 8 * kh;
    out[(size_t)row * Vv + ncol0 + n] = acc[v] + bias;
  }
}

// ---------------------------------------------------------------------------
// Kernel 5: one-pass online log-sum-exp per row (block per row)
// ---------------------------------------------------------------------------
__global__ void lse_kernel(const float* __restrict__ logits,
                           float* __restrict__ lse) {
  __shared__ float smx[256], ssm[256];
  const int r = blockIdx.x;
  const float* rowp = logits + (size_t)r * Vv;
  float mx = -INFINITY, sm = 0.f;
  for (int j = threadIdx.x; j < Vv; j += 256) {
    float v = rowp[j];
    if (v > mx) { sm = sm * expf(mx - v) + 1.f; mx = v; }
    else        { sm += expf(v - mx); }
  }
  smx[threadIdx.x] = mx;
  ssm[threadIdx.x] = sm;
  __syncthreads();
  for (int s = 128; s > 0; s >>= 1) {
    if (threadIdx.x < (unsigned)s) {
      float m1 = smx[threadIdx.x], s1 = ssm[threadIdx.x];
      float m2 = smx[threadIdx.x + s], s2 = ssm[threadIdx.x + s];
      float M = fmaxf(m1, m2);
      float sum = 0.f;
      if (m1 > -INFINITY) sum += s1 * expf(m1 - M);
      if (m2 > -INFINITY) sum += s2 * expf(m2 - M);
      smx[threadIdx.x] = M;
      ssm[threadIdx.x] = sum;
    }
    __syncthreads();
  }
  if (threadIdx.x == 0) lse[r] = smx[0] + logf(ssm[0]);
}

// ---------------------------------------------------------------------------
// Kernel 6: cross entropy at target tokens
// ---------------------------------------------------------------------------
__global__ void ce_kernel(const float* __restrict__ logits,
                          const float* __restrict__ lse,
                          const int* __restrict__ texts,
                          float* __restrict__ ce) {
  int r = blockIdx.x * 256 + threadIdx.x;
  if (r >= Bsz * Tm) return;
  int b = r >> 8, t = r & 255;
  int tgt = texts[b * Tfull + t + 1];
  ce[r] = lse[r] - logits[(size_t)r * Vv + tgt];
}

// ---------------------------------------------------------------------------
// Kernel 7: masked loss reductions + perplexities (65 tail outputs)
// tail layout: loss[16], total_loss[1], full_ppl[16], ending_loss[16], end_ppl[16]
// ---------------------------------------------------------------------------
__global__ void final_kernel(const float* __restrict__ ce,
                             const int* __restrict__ tlen,
                             const int* __restrict__ elen,
                             float* __restrict__ tail) {
  __shared__ float losses[16];
  int b = threadIdx.x;
  if (b < Bsz) {
    int sl = tlen[b] - 1, e = elen[b];
    float s1 = 0.f, s2 = 0.f;
    int n1 = 0, n2 = 0;
    for (int t = 0; t < Tm; ++t) {
      bool mseq = t < sl;
      bool mctx = t < (sl - e);
      float c = ce[b * Tm + t];
      if (mseq) { s1 += c; ++n1; }
      if (mseq && !mctx) { s2 += c; ++n2; }
    }
    float loss = s1 / (float)n1;
    float elo = s2 / (float)n2;
    losses[b] = loss;
    tail[b] = loss;
    tail[17 + b] = expf(loss);
    tail[33 + b] = elo;
    tail[49 + b] = expf(elo);
  }
  __syncthreads();
  if (threadIdx.x == 0) {
    float s = 0.f;
    for (int i = 0; i < Bsz; ++i) s += losses[i];
    tail[16] = s / 16.f;
  }
}

// ---------------------------------------------------------------------------
// Host launcher
// ---------------------------------------------------------------------------
extern "C" void kernel_launch(void* const* d_in, const int* in_sizes, int n_in,
                              void* d_out, int out_size, void* d_ws,
                              size_t ws_size, hipStream_t stream) {
  const int*   texts = (const int*)d_in[0];
  const int*   tlen  = (const int*)d_in[1];
  const int*   elen  = (const int*)d_in[2];
  const float* emb   = (const float*)d_in[3];
  const float* k0    = (const float*)d_in[4];
  const float* b0    = (const float*)d_in[5];
  const float* k1    = (const float*)d_in[6];
  const float* b1    = (const float*)d_in[7];
  const float* Wp    = (const float*)d_in[8];
  const float* bp    = (const float*)d_in[9];

  char* ws = (char*)d_ws;
  int*    bar  = (int*)(ws + WS_BAR);
  __bf16* h0g  = (__bf16*)(ws + WS_H0G);
  __bf16* h1g  = (__bf16*)(ws + WS_H1G);
  __bf16* xbf  = (__bf16*)(ws + WS_XBF);
  __bf16* k0T  = (__bf16*)(ws + WS_K0T);
  __bf16* k1T  = (__bf16*)(ws + WS_K1T);
  __bf16* WpT  = (__bf16*)(ws + WS_WPT);
  __bf16* rnnb = (__bf16*)(ws + WS_RNN);
  float*  lseb = (float*)(ws + WS_LSE);
  float*  ceb  = (float*)(ws + WS_CE);
  float*  logits = (float*)d_out;
  float*  tail   = (float*)d_out + (size_t)Bsz * Tm * Vv;

  // zero barrier + initial hidden state (deterministic every call)
  (void)hipMemsetAsync(ws, 0, WS_ZERO_END, stream);

  embed_kernel<<<4096, 256, 0, stream>>>(texts, emb, xbf);
  transpose_bf16_kernel<<<4096, 256, 0, stream>>>(k0, k0T, Ee + Hh, G4H);
  transpose_bf16_kernel<<<4096, 256, 0, stream>>>(k1, k1T, Hh + Hh, G4H);
  transpose_bf16_kernel<<<4096, 256, 0, stream>>>(Wp, WpT, Hh, Vv);

  (void)hipFuncSetAttribute((const void*)rnn_kernel,
                            hipFuncAttributeMaxDynamicSharedMemorySize,
                            RNN_LDS_BYTES);
  rnn_kernel<<<NBLK_RNN, 128, RNN_LDS_BYTES, stream>>>(
      xbf, k0T, k1T, b0, b1, tlen, h0g, h1g, rnnb, bar);

  logits_kernel<<<dim3(Vv / 128, (Bsz * Tm) / 16), 256, 0, stream>>>(
      rnnb, WpT, bp, logits);
  lse_kernel<<<Bsz * Tm, 256, 0, stream>>>(logits, lseb);
  ce_kernel<<<(Bsz * Tm) / 256, 256, 0, stream>>>(logits, lseb, texts, ceb);
  final_kernel<<<1, 64, 0, stream>>>(ceb, tlen, elen, tail);
}
